// ChannelAttention_57131654971593
// MI455X (gfx1250) — matrix operand order repaired and checked by reading, not compiled, not run on hardware
//
#include <hip/hip_runtime.h>
#include <hip/hip_bf16.h>

#define B_    64
#define IDF   128
#define CDF   256
#define L_    256
#define S2_   1024

typedef __bf16 v16bf __attribute__((ext_vector_type(16)));
typedef float  v8f   __attribute__((ext_vector_type(8)));
typedef int    v4i   __attribute__((vector_size(16)));

#if __has_builtin(__builtin_amdgcn_global_load_async_to_lds_b128)
#define HAVE_ASYNC_LDS 1
typedef __attribute__((address_space(1))) v4i gv4i;   // global v4i
typedef __attribute__((address_space(3))) v4i lv4i;   // LDS v4i
#else
#define HAVE_ASYNC_LDS 0
#endif

#if HAVE_ASYNC_LDS
  #if __has_builtin(__builtin_amdgcn_s_wait_asynccnt)
    #define WAIT_ASYNC(n) __builtin_amdgcn_s_wait_asynccnt(n)
  #else
    #define WAIT_ASYNC(n) asm volatile("s_wait_asynccnt " #n ::: "memory")
  #endif
#else
  #define WAIT_ASYNC(n) ((void)0)
#endif

// LDS B-tile: 64 N-rows x 32 K, row padded to 40 elems (80B = 5x16B) so the 16
// fragment-reading lanes (stride 40 elems) land on distinct bank groups while
// every access stays b128-aligned.
#define BROW 40
#define BTILE (64 * BROW)

// ---- f32 -> bf16 helpers (truncation; 1 v_perm_b32 per packed pair) ---------
__device__ __forceinline__ unsigned pack_bf16x2(float lo, float hi) {
    return __builtin_amdgcn_perm(__builtin_bit_cast(unsigned, hi),
                                 __builtin_bit_cast(unsigned, lo),
                                 0x07060302u);
}
__device__ __forceinline__ __bf16 tobf(float x) {
    unsigned short hs = (unsigned short)(__builtin_bit_cast(unsigned, x) >> 16);
    return __builtin_bit_cast(__bf16, hs);
}

union FragA { v16bf v; uint4 q[2]; unsigned u[8]; };

// A fragment 16x32 from bf16 row-major A: per lane one contiguous 16-elem K-run
// at k0+kh*16  ->  2x global_load_b128.
__device__ __forceinline__ v16bf load_a_bf16(const __bf16* __restrict__ A, int lda,
                                             int row0, int k0, int lane) {
    const int m  = lane & 15;
    const int kh = lane >> 4;
    const uint4* ap = reinterpret_cast<const uint4*>(A + (size_t)(row0 + m) * lda + k0 + kh * 16);
    FragA r;
    r.q[0] = ap[0];   // K: +0..7
    r.q[1] = ap[1];   // K: +8..15
    return r.v;
}

// B fragment 32x16 from the staged LDS tile: per lane 16 contiguous K elems at
// row (t*16+n), byte offset kh*32  ->  2x ds_load_b128.
__device__ __forceinline__ v16bf load_b_lds(const __bf16* sB, int t, int lane) {
    const int n  = lane & 15;
    const int kh = lane >> 4;
    const uint4* p = reinterpret_cast<const uint4*>(sB + (t * 16 + n) * BROW + kh * 16);
    FragA r;
    r.q[0] = p[0];
    r.q[1] = p[1];
    return r.v;
}

// Cooperative stage of one 64(N) x 32(K) bf16 B-tile (4KB): 128 threads x 16B.
// src is column-major [N][K] with leading dim ldb; tile at (col0, k0).
__device__ __forceinline__ void stage_b(__bf16* dst, const __bf16* __restrict__ src,
                                        int ldb, int col0, int k0, int tid) {
    const int row  = tid >> 1;         // 0..63  (N within tile)
    const int half = tid & 1;          // 16-elem half of the K-run
    const __bf16* g = src + (size_t)(col0 + row) * ldb + k0 + half * 16;
    __bf16* l = dst + row * BROW + half * 16;
#if HAVE_ASYNC_LDS
    __builtin_amdgcn_global_load_async_to_lds_b128((gv4i*)g, (lv4i*)l, 0, 0);
#else
    *reinterpret_cast<uint4*>(l) = *reinterpret_cast<const uint4*>(g);
#endif
}

#define WMMA_BF16(a, b, c) \
    __builtin_amdgcn_wmma_f32_16x16x32_bf16(false, (a), false, (b), (short)0, (c), false, false)

// Double-buffered, ASYNCcnt-pipelined GEMM inner loop shared by all 3 GEMMs.
// Each wave owns a 16x64 strip; block = 4 waves along M sharing the B tile.
#define GEMM_LOOP(Aptr, lda, Bptr, ldb, KTOT)                                   \
    v8f acc[4] = {};                                                            \
    stage_b(sB[0], (Bptr), (ldb), col0, 0, threadIdx.x);                        \
    for (int ks = 0; ks < (KTOT) / 32; ++ks) {                                  \
        const int cur = ks & 1;                                                 \
        const int k0  = ks * 32;                                                \
        if (k0 + 32 < (KTOT)) {                                                 \
            stage_b(sB[cur ^ 1], (Bptr), (ldb), col0, k0 + 32, threadIdx.x);    \
            WAIT_ASYNC(2);  /* tile ks and tile ks+1 may both still be in flight */     \
        } else {                                                                \
            WAIT_ASYNC(0);                                                      \
        }                                                                       \
        __syncthreads();                                                        \
        const v16bf a = load_a_bf16((Aptr), (lda), row0, k0, lane);             \
        _Pragma("unroll")                                                       \
        for (int t = 0; t < 4; ++t)                                             \
            acc[t] = WMMA_BF16(a, load_b_lds(sB[cur], t, lane), acc[t]);        \
        __syncthreads();                                                        \
    }

// ---- Pre-pass: f32 -> bf16, same layout (W, weightedContext) ----------------
__global__ void __launch_bounds__(256)
cvt_kernel(const float* __restrict__ in, __bf16* __restrict__ out, int n2) {
    int i = blockIdx.x * 256 + threadIdx.x;
    if (i < n2) {
        float2 f = reinterpret_cast<const float2*>(in)[i];
        reinterpret_cast<unsigned*>(out)[i] = pack_bf16x2(f.x, f.y);
    }
}

// ---- Pre-pass: per-batch transpose + convert: in [R][C] f32 -> out [C][R] bf16
__global__ void __launch_bounds__(256)
tconv_kernel(const float* __restrict__ in, __bf16* __restrict__ out, int R, int C) {
    __shared__ float tile[32][33];
    const float* src = in  + (size_t)blockIdx.z * R * C;
    __bf16*      dst = out + (size_t)blockIdx.z * R * C;
    const int r0 = blockIdx.y * 32, c0 = blockIdx.x * 32;
    const int tx = threadIdx.x & 31, ty = threadIdx.x >> 5;
#pragma unroll
    for (int i = 0; i < 32; i += 8)
        tile[ty + i][tx] = src[(size_t)(r0 + ty + i) * C + c0 + tx];
    __syncthreads();
#pragma unroll
    for (int i = 0; i < 32; i += 8)
        dst[(size_t)(c0 + ty + i) * R + r0 + tx] = tobf(tile[tx][ty + i]);
}

// ---- GEMM1: sourceC[b] = W[S2,CDF] @ context[b][CDF,L] ----------------------
// A = W_bf row-major; B = ctxT column-major [L][CDF]. Writes srcC in BOTH
// row-major [S2][L] (GEMM3 A) and transposed [L][S2] (GEMM2 B).
__global__ void __launch_bounds__(128)
gemm1_kernel(const __bf16* __restrict__ Wb, const __bf16* __restrict__ ctxT,
             __bf16* __restrict__ srcC_rm, __bf16* __restrict__ srcC_t) {
    __shared__ __attribute__((aligned(16))) __bf16 sB[2][BTILE];
    const int lane = threadIdx.x & 31;
    const int wave = threadIdx.x >> 5;
    const int b    = blockIdx.z;
    const int row0 = blockIdx.x * 64 + wave * 16;   // S2
    const int col0 = blockIdx.y * 64;               // L
    const __bf16* Bt = ctxT + (size_t)b * L_ * CDF;

    GEMM_LOOP(Wb, CDF, Bt, CDF, CDF)

    __bf16* orm = srcC_rm + (size_t)b * S2_ * L_;
    __bf16* ot  = srcC_t  + (size_t)b * L_ * S2_;
    const int m_off = (lane >> 4) * 8;
    const int n     = lane & 15;
#pragma unroll
    for (int t = 0; t < 4; ++t)
#pragma unroll
        for (int r = 0; r < 8; ++r) {
            const int s = row0 + r + m_off, l = col0 + t * 16 + n;
            const __bf16 h = tobf(acc[t][r]);
            orm[(size_t)s * L_ + l] = h;
            ot [(size_t)l * S2_ + s] = h;
        }
}

// ---- GEMM2: attn[b] = weightedContext[b][IDF,S2] @ sourceC[b][S2,L] -> f32 --
__global__ void __launch_bounds__(128)
gemm2_kernel(const __bf16* __restrict__ wcb, const __bf16* __restrict__ srcC_t,
             float* __restrict__ attn) {
    __shared__ __attribute__((aligned(16))) __bf16 sB[2][BTILE];
    const int lane = threadIdx.x & 31;
    const int wave = threadIdx.x >> 5;
    const int b    = blockIdx.z;
    const int row0 = blockIdx.x * 64 + wave * 16;   // IDF
    const int col0 = blockIdx.y * 64;               // L
    const __bf16* A  = wcb    + (size_t)b * IDF * S2_;
    const __bf16* Bt = srcC_t + (size_t)b * L_ * S2_;

    GEMM_LOOP(A, S2_, Bt, S2_, S2_)

    float* out = attn + (size_t)b * IDF * L_;
    const int m_off = (lane >> 4) * 8;
    const int n     = lane & 15;
#pragma unroll
    for (int t = 0; t < 4; ++t)
#pragma unroll
        for (int r = 0; r < 8; ++r)
            out[(size_t)(row0 + r + m_off) * L_ + col0 + t * 16 + n] = acc[t][r];
}

// ---- Softmax over L per (b,i) row; emits bf16 [IDF][L] and f32 [B,L,IDF] ----
__global__ void __launch_bounds__(256)
softmax_kernel(const float* __restrict__ attn, __bf16* __restrict__ attn_bf,
               float* __restrict__ out2) {
    __shared__ float red[256];
    const int row = blockIdx.x;        // b*IDF + i
    const int t   = threadIdx.x;       // L index
    const float x = attn[(size_t)row * L_ + t];

    red[t] = x; __syncthreads();
#pragma unroll
    for (int s = 128; s > 0; s >>= 1) {
        if (t < s) red[t] = fmaxf(red[t], red[t + s]);
        __syncthreads();
    }
    const float mx = red[0]; __syncthreads();

    const float e = __expf(x - mx);
    red[t] = e; __syncthreads();
#pragma unroll
    for (int s = 128; s > 0; s >>= 1) {
        if (t < s) red[t] += red[t + s];
        __syncthreads();
    }
    const float p = e / red[0];

    attn_bf[(size_t)row * L_ + t] = tobf(p);
    const int b = row / IDF, i = row % IDF;
    out2[((size_t)b * L_ + t) * IDF + i] = p;   // attn_c [B, L, IDF]
}

// ---- GEMM3: wcT[b] = sourceC[b][S2,L] @ attn[b]^T[L,IDF] -> out1 [B,IDF,S2] -
__global__ void __launch_bounds__(128)
gemm3_kernel(const __bf16* __restrict__ srcC_rm, const __bf16* __restrict__ attn_bf,
             float* __restrict__ out1) {
    __shared__ __attribute__((aligned(16))) __bf16 sB[2][BTILE];
    const int lane = threadIdx.x & 31;
    const int wave = threadIdx.x >> 5;
    const int b    = blockIdx.z;
    const int row0 = blockIdx.x * 64 + wave * 16;   // S2
    const int col0 = blockIdx.y * 64;               // IDF
    const __bf16* A  = srcC_rm + (size_t)b * S2_ * L_;
    const __bf16* Bt = attn_bf + (size_t)b * IDF * L_;

    GEMM_LOOP(A, L_, Bt, L_, L_)

    float* out = out1 + (size_t)b * IDF * S2_;
    const int m_off = (lane >> 4) * 8;
    const int n     = lane & 15;
#pragma unroll
    for (int t = 0; t < 4; ++t)
#pragma unroll
        for (int r = 0; r < 8; ++r)   // out[b][i][s] = wcT[s][i]
            out[(size_t)(col0 + t * 16 + n) * S2_ + row0 + r + m_off] = acc[t][r];
}

extern "C" void kernel_launch(void* const* d_in, const int* in_sizes, int n_in,
                              void* d_out, int out_size, void* d_ws, size_t ws_size,
                              hipStream_t stream) {
    const float* wc  = (const float*)d_in[0];   // [B, IDF, S2]
    const float* ctx = (const float*)d_in[1];   // [B, CDF, L]
    const float* W   = (const float*)d_in[2];   // [S2, CDF]

    float* out1 = (float*)d_out;                 // wcT  : B*IDF*S2
    float* out2 = out1 + (size_t)B_ * IDF * S2_; // attn : B*L*IDF

    // ---- workspace layout (byte offsets) -----------------------------------
    char* ws = (char*)d_ws;
    float*  attn_f   = (float*)ws;                             //  8 MB  [B][IDF][L] f32
    __bf16* srcC_rm  = (__bf16*)(ws + (((size_t) 8) << 20));   // 32 MB  [B][S2][L]
    __bf16* srcC_t   = (__bf16*)(ws + (((size_t)40) << 20));   // 32 MB  [B][L][S2]
    __bf16* ctxT     = (__bf16*)(ws + (((size_t)72) << 20));   //  8 MB  [B][L][CDF]
    __bf16* wcb      = (__bf16*)(ws + (((size_t)80) << 20));   // 16 MB  [B][IDF][S2]
    __bf16* Wb       = (__bf16*)(ws + (((size_t)96) << 20));   // 0.5MB  [S2][CDF]
    __bf16* attn_bf  = (__bf16*)(ws + (((size_t)97) << 20));   //  4 MB  [B][IDF][L]

    // pre-passes: convert operands to bf16 in fragment-friendly layouts
    cvt_kernel<<<dim3((S2_ * CDF / 2) / 256), 256, 0, stream>>>(W, Wb, S2_ * CDF / 2);
    cvt_kernel<<<dim3((B_ * IDF * S2_ / 2) / 256), 256, 0, stream>>>(wc, wcb, B_ * IDF * S2_ / 2);
    tconv_kernel<<<dim3(L_ / 32, CDF / 32, B_), 256, 0, stream>>>(ctx, ctxT, CDF, L_);

    gemm1_kernel<<<dim3(S2_ / 64, L_ / 64, B_), 128, 0, stream>>>(Wb, ctxT, srcC_rm, srcC_t);
    gemm2_kernel<<<dim3(IDF / 64, L_ / 64, B_), 128, 0, stream>>>(wcb, srcC_t, attn_f);
    softmax_kernel<<<dim3(B_ * IDF), 256, 0, stream>>>(attn_f, attn_bf, out2);
    gemm3_kernel<<<dim3(S2_ / 64, IDF / 64, B_), 128, 0, stream>>>(srcC_rm, attn_bf, out1);
}
